// Lag_PP_zero_5970004542131
// MI455X (gfx1250) — compile-verified, matrix operand order checked by reading, not run on hardware
//
#include <hip/hip_runtime.h>

#define LL 512
#define BB 16
#define TS 64
#define NT 8                 // tiles per dim (512/64)
#define NPAIR 36             // NT*(NT+1)/2
#define TSTEPS 20
#define SCONST 2.1972245773362196f   // log(9)

typedef __attribute__((ext_vector_type(2)))  float    v2f;
typedef __attribute__((ext_vector_type(8)))  float    v8f;
typedef __attribute__((ext_vector_type(16))) _Float16 v16h;

__device__ __forceinline__ float sigm(float x) { return 1.0f / (1.0f + __expf(-x)); }

__device__ __forceinline__ void pair_decode(int p, int& it, int& jt) {
  it = 0;
  while (p >= NT - it) { p -= NT - it; ++it; }
  jt = it + p;
}

// Row-sums (transposed==0) or col-sums (transposed==1) of a 16-row slab (group g)
// of a 64x64 LDS tile (row stride 65) via the matrix pipe:
//   D = A_slab(16x4) x ones(4x16), chained over 16 K-chunks.
// Every column of D equals the row sums; lanes 0 and 16 write them out.
// pdst points at partial[(b*LL + rowBase)*NT + slot]; row stride NT.
__device__ __forceinline__ void slab_reduce_wmma(const float* __restrict__ ans,
                                                 int g, int transposed,
                                                 float* __restrict__ pdst)
{
  const int lane = threadIdx.x & 31;
  const int m    = lane & 15;
  v8f acc = {0.f, 0.f, 0.f, 0.f, 0.f, 0.f, 0.f, 0.f};
#if __has_builtin(__builtin_amdgcn_wmma_f32_16x16x4_f32)
  const int koff = (lane >> 4) << 1;    // A f32 16x4 layout: lanes16-31 hold K+2/K+3
  v2f ones; ones[0] = 1.0f; ones[1] = 1.0f;
  #pragma unroll
  for (int kk = 0; kk < 16; ++kk) {
    const int k0 = kk * 4 + koff;
    v2f a;
    if (!transposed) {
      a[0] = ans[(g * 16 + m) * 65 + k0];
      a[1] = ans[(g * 16 + m) * 65 + k0 + 1];
    } else {
      a[0] = ans[(k0    ) * 65 + g * 16 + m];
      a[1] = ans[(k0 + 1) * 65 + g * 16 + m];
    }
    acc = __builtin_amdgcn_wmma_f32_16x16x4_f32(false, a, false, ones,
                                                (short)0, acc, false, false);
  }
#else
  // Fallback: f16 WMMA (codegen-confirmed builtin), K=32 chunks over the 64 cols.
  v16h hones;
  #pragma unroll
  for (int e = 0; e < 16; ++e) hones[e] = (_Float16)1.0f;
  #pragma unroll
  for (int kk = 0; kk < 2; ++kk) {
    v16h av;
    #pragma unroll
    for (int e = 0; e < 16; ++e) {
      const int K = kk * 32 + (e & 7) + ((e >> 3) << 4) + ((lane >> 4) << 3);
      const float v = transposed ? ans[K * 65 + g * 16 + m]
                                 : ans[(g * 16 + m) * 65 + K];
      av[e] = (_Float16)v;
    }
    acc = __builtin_amdgcn_wmma_f32_16x16x32_f16(false, av, false, hones,
                                                 (short)0, acc, false, false);
  }
#endif
  if (m == 0) {                         // lane 0 -> rows 0..7, lane 16 -> rows 8..15
    const int rbase = (lane >> 4) * 8;
    #pragma unroll
    for (int q = 0; q < 8; ++q) pdst[(size_t)(rbase + q) * NT] = acc[q];
  }
}

// ---------------------------------------------------------------------------
// Init: m = constraint_matrix(x); u' = softsign(u-s)*u; a_hat0; rowsum(a0) partials.
// One WG per (batch, tile-pair it<=jt).
// ---------------------------------------------------------------------------
__global__ void __launch_bounds__(256) init_kernel(
    const float* __restrict__ u, const float* __restrict__ x,
    float* __restrict__ up, float* __restrict__ ah, float* __restrict__ mbuf,
    float* __restrict__ partial)
{
  __shared__ float sAhM[TS][65];
  __shared__ float sAns[TS][65];
  __shared__ float sXi[TS][4], sXj[TS][4];
  const int b = blockIdx.y;
  int it, jt; pair_decode(blockIdx.x, it, jt);
  const int tid = threadIdx.x, l16 = tid & 15, rq = tid >> 4;
  const size_t baseT = ((size_t)b * LL + (size_t)it * TS) * LL + (size_t)jt * TS;
  const size_t baseM = ((size_t)b * LL + (size_t)jt * TS) * LL + (size_t)it * TS;

  if (tid < TS) {
    float4 xv = *(const float4*)(x + ((size_t)b * LL + (size_t)it * TS + tid) * 4);
    sXi[tid][0] = xv.x; sXi[tid][1] = xv.y; sXi[tid][2] = xv.z; sXi[tid][3] = xv.w;
    float4 yv = *(const float4*)(x + ((size_t)b * LL + (size_t)jt * TS + tid) * 4);
    sXj[tid][0] = yv.x; sXj[tid][1] = yv.y; sXj[tid][2] = yv.z; sXj[tid][3] = yv.w;
  }
  __syncthreads();

  float ahT[4][4], mT[4][4];
  #pragma unroll
  for (int pp = 0; pp < 4; ++pp) {
    const int r = rq + 16 * pp, c = l16 * 4;
    float4 utv = *(const float4*)(u + baseT + (size_t)r * LL + c);
    float4 umv = *(const float4*)(u + baseM + (size_t)r * LL + c);
    float4 upT4, upM4, ahT4, ahM4, mT4, mM4;
    #pragma unroll
    for (int q = 0; q < 4; ++q) {
      const int cc = c + q;
      // straight element (it*TS+r, jt*TS+cc)
      {
        float uv  = ((const float*)&utv)[q];
        float upv = sigm(2.f * (uv - SCONST)) * uv;
        float ahv = sigm(upv) * sigm(2.f * (upv - SCONST));
        const float* xi = sXi[r]; const float* xj = sXj[cc];
        float mv = xi[0]*xj[1] + xj[0]*xi[1] + xi[2]*xj[3] + xj[2]*xi[3]
                 + xi[1]*xj[3] + xj[1]*xi[3];
        int di = (it * TS + r) - (jt * TS + cc);
        if (di <= 3 && di >= -3) mv = 0.f;
        ((float*)&upT4)[q] = upv; ((float*)&ahT4)[q] = ahv; ((float*)&mT4)[q] = mv;
        ahT[pp][q] = ahv; mT[pp][q] = mv;
      }
      // mirror element (jt*TS+r, it*TS+cc)
      {
        float uv  = ((const float*)&umv)[q];
        float upv = sigm(2.f * (uv - SCONST)) * uv;
        float ahv = sigm(upv) * sigm(2.f * (upv - SCONST));
        const float* xi = sXj[r]; const float* xj = sXi[cc];
        float mv = xi[0]*xj[1] + xj[0]*xi[1] + xi[2]*xj[3] + xj[2]*xi[3]
                 + xi[1]*xj[3] + xj[1]*xi[3];
        int di = (jt * TS + r) - (it * TS + cc);
        if (di <= 3 && di >= -3) mv = 0.f;
        ((float*)&upM4)[q] = upv; ((float*)&ahM4)[q] = ahv; ((float*)&mM4)[q] = mv;
        sAhM[r][cc] = ahv;
      }
    }
    *(float4*)(up   + baseT + (size_t)r * LL + c) = upT4;
    *(float4*)(ah   + baseT + (size_t)r * LL + c) = ahT4;
    *(float4*)(mbuf + baseT + (size_t)r * LL + c) = mT4;
    if (it != jt) {
      *(float4*)(up   + baseM + (size_t)r * LL + c) = upM4;
      *(float4*)(ah   + baseM + (size_t)r * LL + c) = ahM4;
      *(float4*)(mbuf + baseM + (size_t)r * LL + c) = mM4;
    }
  }
  __syncthreads();

  #pragma unroll
  for (int pp = 0; pp < 4; ++pp) {
    const int r = rq + 16 * pp, c = l16 * 4;
    #pragma unroll
    for (int q = 0; q < 4; ++q) {
      const int cc = c + q;
      float a1 = ahT[pp][q], a2 = sAhM[cc][r];
      sAns[r][cc] = 0.5f * (a1 * a1 + a2 * a2) * mT[pp][q];
    }
  }
  __syncthreads();

  const int w = tid >> 5;
  if (w < 4) {
    slab_reduce_wmma(&sAns[0][0], w, 0,
        partial + (size_t)(b * LL + it * TS + w * 16) * NT + jt);
  } else if (it != jt) {
    slab_reduce_wmma(&sAns[0][0], w - 4, 1,
        partial + (size_t)(b * LL + jt * TS + (w - 4) * 16) * NT + it);
  }
}

// ---------------------------------------------------------------------------
// One Lagrangian step: a_hat update + a_new (written to the output stack) +
// deterministic per-tile row-sum partials via WMMA.
// ---------------------------------------------------------------------------
__global__ void __launch_bounds__(256) step_kernel(
    const float* __restrict__ up, const float* __restrict__ mbuf,
    float* __restrict__ ah, const float* __restrict__ lam,
    float* __restrict__ outp, float* __restrict__ partial,
    float alpha, float thresh)
{
  __shared__ float sUt[TS][65];    // u' straight tile; reused as a_new after phase 2
  __shared__ float sUm[TS][65];    // u' mirror tile
  __shared__ float sAhnM[TS][65];  // updated a_hat, mirror tile
  __shared__ float sLi[TS], sLj[TS];
  const int b = blockIdx.y;
  int it, jt; pair_decode(blockIdx.x, it, jt);
  const int tid = threadIdx.x, l16 = tid & 15, rq = tid >> 4;
  const size_t baseT = ((size_t)b * LL + (size_t)it * TS) * LL + (size_t)jt * TS;
  const size_t baseM = ((size_t)b * LL + (size_t)jt * TS) * LL + (size_t)it * TS;

  if (tid < TS) {
    sLi[tid] = lam[b * LL + it * TS + tid];
    sLj[tid] = lam[b * LL + jt * TS + tid];
  }
  #pragma unroll
  for (int pp = 0; pp < 4; ++pp) {
    const int r = rq + 16 * pp, c = l16 * 4;
    float4 utv = *(const float4*)(up + baseT + (size_t)r * LL + c);
    float4 umv = *(const float4*)(up + baseM + (size_t)r * LL + c);
    sUt[r][c] = utv.x; sUt[r][c+1] = utv.y; sUt[r][c+2] = utv.z; sUt[r][c+3] = utv.w;
    sUm[r][c] = umv.x; sUm[r][c+1] = umv.y; sUm[r][c+2] = umv.z; sUm[r][c+3] = umv.w;
  }
  __syncthreads();

  float ahnT[4][4], mT[4][4];
  #pragma unroll
  for (int pp = 0; pp < 4; ++pp) {
    const int r = rq + 16 * pp, c = l16 * 4;
    float4 atv = *(const float4*)(ah   + baseT + (size_t)r * LL + c);
    float4 amv = *(const float4*)(ah   + baseM + (size_t)r * LL + c);
    float4 mtv = *(const float4*)(mbuf + baseT + (size_t)r * LL + c);
    float4 mmv = *(const float4*)(mbuf + baseM + (size_t)r * LL + c);
    float4 ahnT4, ahnM4;
    #pragma unroll
    for (int q = 0; q < 4; ++q) {
      const int cc = c + q;
      // straight: S = -0.5*(u'[i,j] + u'[j,i]) + lam[i] + lam[j]
      float S  = -0.5f * (sUt[r][cc] + sUm[cc][r]) + sLi[r] + sLj[cc];
      float a0 = ((const float*)&atv)[q];
      float mv = ((const float*)&mtv)[q];
      float v  = a0 - alpha * (a0 * mv * S);
      v = fminf(fmaxf(fabsf(v) - thresh, 0.f), 1.f);
      ahnT[pp][q] = v; mT[pp][q] = mv;
      ((float*)&ahnT4)[q] = v;
      // mirror
      float Sm  = -0.5f * (sUm[r][cc] + sUt[cc][r]) + sLj[r] + sLi[cc];
      float am0 = ((const float*)&amv)[q];
      float mm  = ((const float*)&mmv)[q];
      float vm  = am0 - alpha * (am0 * mm * Sm);
      vm = fminf(fmaxf(fabsf(vm) - thresh, 0.f), 1.f);
      ((float*)&ahnM4)[q] = vm;
      sAhnM[r][cc] = vm;
    }
    *(float4*)(ah + baseT + (size_t)r * LL + c) = ahnT4;
    if (it != jt) *(float4*)(ah + baseM + (size_t)r * LL + c) = ahnM4;
  }
  __syncthreads();

  float (*sAns)[65] = sUt;   // u' tiles no longer needed; reuse as a_new store
  #pragma unroll
  for (int pp = 0; pp < 4; ++pp) {
    const int r = rq + 16 * pp, c = l16 * 4;
    float4 an;
    #pragma unroll
    for (int q = 0; q < 4; ++q) {
      const int cc = c + q;
      float a1 = ahnT[pp][q], a2 = sAhnM[cc][r];
      float v = 0.5f * (a1 * a1 + a2 * a2) * mT[pp][q];
      ((float*)&an)[q] = v;
      sAns[r][cc] = v;
    }
    *(float4*)(outp + baseT + (size_t)r * LL + c) = an;
  }
  __syncthreads();

  if (it != jt) {   // a is symmetric: mirror block is the transpose
    #pragma unroll
    for (int pp = 0; pp < 4; ++pp) {
      const int r = rq + 16 * pp, c = l16 * 4;
      float4 an;
      #pragma unroll
      for (int q = 0; q < 4; ++q) ((float*)&an)[q] = sAns[c + q][r];
      *(float4*)(outp + baseM + (size_t)r * LL + c) = an;
    }
  }

  const int w = tid >> 5;
  if (w < 4) {
    slab_reduce_wmma(&sAns[0][0], w, 0,
        partial + (size_t)(b * LL + it * TS + w * 16) * NT + jt);
  } else if (it != jt) {
    slab_reduce_wmma(&sAns[0][0], w - 4, 1,
        partial + (size_t)(b * LL + jt * TS + (w - 4) * 16) * NT + it);
  }
}

// ---------------------------------------------------------------------------
// Finalize: rs = sum of 8 deterministic partials; lmbd update; lam = lmbd*sig(2(rs-1)).
// ---------------------------------------------------------------------------
__global__ void __launch_bounds__(256) finalize_kernel(
    const float* __restrict__ partial, float* __restrict__ lmbd,
    float* __restrict__ lam, float beta, int initFlag)
{
  const int i = blockIdx.x * 256 + threadIdx.x;
  if (i >= BB * LL) return;
  float rs = 0.f;
  #pragma unroll
  for (int s = 0; s < NT; ++s) rs += partial[(size_t)i * NT + s];
  float prev = initFlag ? 0.f : lmbd[i];
  float l = prev + beta * fmaxf(rs - 1.f, 0.f);
  lmbd[i] = l;
  lam[i]  = l * sigm(2.f * (rs - 1.f));
}

extern "C" void kernel_launch(void* const* d_in, const int* in_sizes, int n_in,
                              void* d_out, int out_size, void* d_ws, size_t ws_size,
                              hipStream_t stream)
{
  (void)in_sizes; (void)n_in; (void)out_size;
  const float* u = (const float*)d_in[0];
  const float* x = (const float*)d_in[1];
  const size_t NBLL = (size_t)BB * LL * LL;
  const size_t need = (3 * NBLL + (size_t)BB * LL * NT + 2 * (size_t)BB * LL) * sizeof(float);
  if (ws_size < need) return;

  float* ws      = (float*)d_ws;
  float* mbuf    = ws;
  float* up      = mbuf + NBLL;
  float* ah      = up + NBLL;
  float* partial = ah + NBLL;
  float* lmbd    = partial + (size_t)BB * LL * NT;
  float* lam     = lmbd + (size_t)BB * LL;
  float* outp    = (float*)d_out;

  dim3 grid(NPAIR, BB), blk(256);
  dim3 fgrid((BB * LL + 255) / 256);

  init_kernel<<<grid, blk, 0, stream>>>(u, x, up, ah, mbuf, partial);
  finalize_kernel<<<fgrid, blk, 0, stream>>>(partial, lmbd, lam, 1.0f, 1);

  float alpha = 0.01f;   // ALPHA0 * DECAY^t
  float beta  = 0.1f;    // BETA0  * DECAY^t
  for (int t = 0; t < TSTEPS; ++t) {
    step_kernel<<<grid, blk, 0, stream>>>(up, mbuf, ah, lam,
                                          outp + (size_t)t * NBLL, partial,
                                          alpha, alpha * 0.99f /* RHO*alpha*DECAY */);
    finalize_kernel<<<fgrid, blk, 0, stream>>>(partial, lmbd, lam, beta, 0);
    alpha *= 0.99f;
    beta  *= 0.99f;
  }
}
